// acRKN_48412871360736
// MI455X (gfx1250) — matrix operand order repaired and checked by reading, not compiled, not run on hardware
//
#include <hip/hip_runtime.h>
#include <hip/hip_bf16.h>

// ---------------------------------------------------------------------------
// acRKN forward for MI455X (gfx1250).
//  - Batch GEMM phases (encoder/ctrl/decoders): bf16 WMMA 16x16x32, f32 accum.
//  - Kalman scan: fp32 VALU, banded (7-diagonal) transition algebra,
//    register-resident state, async-to-LDS double-buffered input prefetch,
//    basis/coeff tables staged in LDS.
// ---------------------------------------------------------------------------

#define B_DIM 512
#define T_DIM 300
#define BT    (B_DIM * T_DIM)
#define LOD   64
#define LSD   128
#define NB    15

#define BB_ELEMS (4 * LOD * 7 * NB)   // 26880 banded-basis floats
#define WC_ELEMS (128 * NB)           // 1920 coeff-weight floats

typedef __bf16 v16bf __attribute__((ext_vector_type(16)));
typedef float  f32x8 __attribute__((ext_vector_type(8)));

union Frag {
    v16bf    v;
    unsigned u[8];
};

__device__ __forceinline__ unsigned short f2bf(float f) {
    unsigned u = __builtin_bit_cast(unsigned, f);
    u += 0x7FFFu + ((u >> 16) & 1u);          // round-to-nearest-even
    return (unsigned short)(u >> 16);
}

__device__ __forceinline__ float elup1f(float x) {
    return x > 0.f ? x + 1.f : __expf(x);
}

// -------------------- weight pre-swizzle into WMMA B-fragment layout --------
// B matrix 32x16 bf16 per wave: VGPR v, lane l: n = 16*tile + (l&15),
// K = 32*kc + (v>>2)*16 + (l>>4)*8 + 2*(v&3) (+1 in high half of the dword).
__global__ void prep_bfrag(const float* __restrict__ W, int Ks, int Ns,
                           int Kp, int Np, unsigned* __restrict__ dst) {
    int nk = Kp >> 5;
    int total = (Np >> 4) * nk * 32 * 8;
    for (int idx = blockIdx.x * blockDim.x + threadIdx.x; idx < total;
         idx += gridDim.x * blockDim.x) {
        int v     = idx & 7;
        int lane  = (idx >> 3) & 31;
        int rest  = idx >> 8;
        int kc    = rest % nk;
        int ntile = rest / nk;
        int n = ntile * 16 + (lane & 15);
        int K = kc * 32 + (v >> 2) * 16 + (lane >> 4) * 8 + (v & 3) * 2;
        float lo = (K     < Ks && n < Ns) ? W[(size_t)K       * Ns + n] : 0.f;
        float hi = (K + 1 < Ks && n < Ns) ? W[(size_t)(K + 1) * Ns + n] : 0.f;
        dst[idx] = (unsigned)f2bf(lo) | ((unsigned)f2bf(hi) << 16);
    }
}

// -------------------- banded basis extraction: bb[m][i][dd][k] --------------
__global__ void prep_basis(const float* __restrict__ basis, float* __restrict__ bb) {
    int idx = blockIdx.x * blockDim.x + threadIdx.x;
    if (idx >= BB_ELEMS) return;
    int k  = idx % NB;
    int dd = (idx / NB) % 7;
    int i  = (idx / (NB * 7)) % LOD;
    int m  = idx / (NB * 7 * LOD);
    int j  = i + dd - 3;
    float v = (j >= 0 && j < LOD)
                  ? basis[(((size_t)m * NB + k) * LOD + i) * LOD + j]
                  : 0.f;
    bb[idx] = v;
}

// -------------------- WMMA tile helpers (one wave = one 16-row tile) --------
__device__ __forceinline__ void load_a_frag(const unsigned short* sA, int Kp,
                                            int kbase, Frag& a) {
    int lane = threadIdx.x & 31;
    int m = lane & 15, hf = lane >> 4;
    const unsigned* p = (const unsigned*)sA;
#pragma unroll
    for (int v = 0; v < 8; ++v) {
        int K0 = (v >> 2) * 16 + (v & 3) * 2 + hf * 8;  // ISA 16-bit A layout
        a.u[v] = p[(m * Kp + kbase + K0) >> 1];
    }
}

__device__ __forceinline__ void load_b_frag(const unsigned* wf, int ntile,
                                            int nk, int kc, Frag& b) {
    int lane = threadIdx.x & 31;
    const unsigned* p = wf + ((size_t)(ntile * nk + kc) * 32 + lane) * 8;
#pragma unroll
    for (int v = 0; v < 8; ++v) b.u[v] = p[v];
}

// X[16 x Ksrc] f32 (global, row-major) -> LDS bf16 [16 x Kp] zero-padded.
__device__ __forceinline__ void stage_tile(const float* src, int row0, int Ksrc,
                                           int Kp, unsigned short* dst) {
    for (int e = threadIdx.x; e < 16 * Kp; e += 32) {
        int r = e / Kp, c = e % Kp;
        float v = (c < Ksrc) ? src[(size_t)(row0 + r) * Ksrc + c] : 0.f;
        dst[e] = f2bf(v);
    }
}

// Hidden layer: LDS bf16 A -> WMMA -> bias(+relu) -> LDS bf16 D.
__device__ __forceinline__ void layer_lds(const unsigned short* sA, int Kp,
                                          const unsigned* wf, const float* bias,
                                          int Np, unsigned short* sD, bool relu) {
    int lane = threadIdx.x & 31;
    int m0 = lane & 15, hf = lane >> 4;
    int nk = Kp >> 5, nt = Np >> 4;
    for (int ntile = 0; ntile < nt; ++ntile) {
        f32x8 acc;
#pragma unroll
        for (int r = 0; r < 8; ++r) acc[r] = 0.f;
        for (int kc = 0; kc < nk; ++kc) {
            Frag a, b;
            load_a_frag(sA, Kp, kc * 32, a);
            load_b_frag(wf, ntile, nk, kc, b);
            acc = __builtin_amdgcn_wmma_f32_16x16x32_bf16(
                false, a.v, false, b.v, (short)0, acc, false, false);
        }
        int n = ntile * 16 + m0;
        float bv = bias[n];
#pragma unroll
        for (int r = 0; r < 8; ++r) {
            float x = acc[r] + bv;
            if (relu) x = x > 0.f ? x : 0.f;
            sD[(r + hf * 8) * Np + n] = f2bf(x);
        }
    }
}

// Output layer: LDS bf16 A -> WMMA -> bias + act(mode) -> f32 global.
// mode: 0=none, 1=relu, 2=elup1.
__device__ __forceinline__ void layer_out(const unsigned short* sA, int Kp,
                                          const unsigned* wf, const float* bias,
                                          int Np, float* dst, int ld, int Nvalid,
                                          int row0, int mode) {
    int lane = threadIdx.x & 31;
    int m0 = lane & 15, hf = lane >> 4;
    int nk = Kp >> 5, nt = Np >> 4;
    for (int ntile = 0; ntile < nt; ++ntile) {
        f32x8 acc;
#pragma unroll
        for (int r = 0; r < 8; ++r) acc[r] = 0.f;
        for (int kc = 0; kc < nk; ++kc) {
            Frag a, b;
            load_a_frag(sA, Kp, kc * 32, a);
            load_b_frag(wf, ntile, nk, kc, b);
            acc = __builtin_amdgcn_wmma_f32_16x16x32_bf16(
                false, a.v, false, b.v, (short)0, acc, false, false);
        }
        int n = ntile * 16 + m0;
        if (n >= Nvalid) continue;
        float bv = bias[n];
#pragma unroll
        for (int r = 0; r < 8; ++r) {
            float x = acc[r] + bv;
            if (mode == 1) x = x > 0.f ? x : 0.f;
            if (mode == 2) x = elup1f(x);
            dst[(size_t)(row0 + r + hf * 8) * ld + n] = x;
        }
    }
}

// -------------------- kernel A: encoder + ctrl MLPs -------------------------
__global__ __launch_bounds__(32) void enc_kernel(
    const float* __restrict__ obs, const float* __restrict__ act,
    const unsigned* wfE0, const float* bE0, const unsigned* wfE1, const float* bE1,
    const unsigned* wfEM, const float* bEM, const unsigned* wfEV, const float* bEV,
    const unsigned* wfC0, const float* bC0, const unsigned* wfC1, const float* bC1,
    float* __restrict__ obsm, float* __restrict__ obsv, float* __restrict__ ctrlv) {
    __shared__ unsigned sXu[16 * 192 / 2];
    __shared__ unsigned sHu[16 * 128 / 2];
    unsigned short* sX = (unsigned short*)sXu;
    unsigned short* sH = (unsigned short*)sHu;
    int row0 = blockIdx.x * 16;

    stage_tile(obs, row0, 60, 64, sX);          __syncthreads();
    layer_lds(sX, 64, wfE0, bE0, 128, sH, true); __syncthreads();
    layer_lds(sH, 128, wfE1, bE1, 128, sX, true); __syncthreads();
    layer_out(sX, 128, wfEM, bEM, 64, obsm, 64, 64, row0, 0);
    layer_out(sX, 128, wfEV, bEV, 64, obsv, 64, 64, row0, 2);
    __syncthreads();
    stage_tile(act, row0, 12, 32, sH);          __syncthreads();
    layer_lds(sH, 32, wfC0, bC0, 128, sX, true); __syncthreads();
    layer_out(sX, 128, wfC1, bC1, 128, ctrlv, 128, 128, row0, 0);
}

// -------------------- kernel B: sequential Kalman scan ----------------------
// 64 lanes per batch row (lane = state index i), 4 rows per 256-thread block.
// Inputs for step t+1 are prefetched with GLOBAL_LOAD_ASYNC_TO_LDS_B128 while
// step t computes; basis + coeff weights live in LDS.
__global__ __launch_bounds__(256) void scan_kernel(
    const float* __restrict__ obsm, const float* __restrict__ obsv,
    const float* __restrict__ ctrlv, const unsigned char* __restrict__ validp,
    const float* __restrict__ Wc, const float* __restrict__ bc,
    const float* __restrict__ bb, const float* __restrict__ lpn,
    float* __restrict__ pm_out, float* __restrict__ pc_out) {
    int r = threadIdx.x >> 6;
    int i = threadIdx.x & 63;
    int b = blockIdx.x * 4 + r;

    __shared__ float s_pm[4][128];
    __shared__ float s_cu[4][64], s_cl[4][64], s_cs[4][64];
    __shared__ float s_cr[4][16];
    __shared__ __align__(16) float s_in[2][4][256];  // om[64]|ov[64]|u[128]

    extern __shared__ float dynsm[];
    float* s_bb = dynsm;                  // BB_ELEMS
    float* s_wc = dynsm + BB_ELEMS;       // WC_ELEMS
    float* s_cb = s_wc + WC_ELEMS;        // 16

    // async prefetch of one timestep's inputs for this row into s_in[buf]
    auto issue = [&](int t, int buf) {
        size_t rt = (size_t)b * T_DIM + t;
        const float* src;
        unsigned dsti;
        if (i < 16)      { src = obsm  + rt * 64  + i * 4;        dsti = i * 4; }
        else if (i < 32) { src = obsv  + rt * 64  + (i - 16) * 4; dsti = 64 + (i - 16) * 4; }
        else             { src = ctrlv + rt * 128 + (i - 32) * 4; dsti = 128 + (i - 32) * 4; }
        unsigned ldsoff = (unsigned)(unsigned long long)(const void*)&s_in[buf][r][dsti];
        asm volatile("global_load_async_to_lds_b128 %0, %1, off"
                     :: "v"(ldsoff), "v"(src) : "memory");
    };

    issue(0, 0);  // prefetch t=0 while we stage the tables

    // stage banded basis + coeff weights into LDS (block-cooperative)
    for (int e = threadIdx.x; e < BB_ELEMS; e += 256) s_bb[e] = bb[e];
    for (int e = threadIdx.x; e < WC_ELEMS; e += 256) s_wc[e] = Wc[e];
    if (threadIdx.x < NB) s_cb[threadIdx.x] = bc[threadIdx.x];

    float noise_u = elup1f(lpn[i]);
    float noise_l = elup1f(lpn[64 + i]);
    float mu_u = 0.f, mu_l = 0.f, cu = 10.f, cl = 10.f, cs = 1.f;

    for (int t = 0; t < T_DIM; ++t) {
        if (t + 1 < T_DIM) {
            issue(t + 1, (t + 1) & 1);
            asm volatile("s_wait_asynccnt 0x1" ::: "memory");  // step t ready
        } else {
            asm volatile("s_wait_asynccnt 0x0" ::: "memory");
        }
        __syncthreads();  // cross-wave visibility of s_in (and staged tables)

        size_t rt = (size_t)b * T_DIM + t;
        const float* inb = s_in[t & 1][r];
        float om = inb[i];
        float ov = inb[64 + i];
        float uu = inb[128 + i];
        float ul = inb[192 + i];
        bool valid = validp[rt] != 0;

        // Kalman update (per-lane, registers only)
        float denom = cu + ov;
        float qu = cu / denom, ql = cs / denom;
        float res = om - mu_u;
        float pmu = valid ? mu_u + qu * res : mu_u;
        float pml = valid ? mu_l + ql * res : mu_l;
        float pcu = valid ? (1.f - qu) * cu : cu;
        float pcl = valid ? cl - ql * cs : cl;
        float pcs = valid ? (1.f - qu) * cs : cs;

        s_pm[r][i] = pmu; s_pm[r][64 + i] = pml;
        s_cu[r][i] = pcu; s_cl[r][i] = pcl; s_cs[r][i] = pcs;
        __syncthreads();

        // coefficient MLP 128->15 (15 lanes do full dots from LDS; tiny)
        if (i < NB) {
            float a = s_cb[i];
            for (int k = 0; k < 128; ++k) a += s_pm[r][k] * s_wc[k * NB + i];
            s_cr[r][i] = a;
        }
        __syncthreads();

        // redundant per-lane softmax over 15 (cheap, avoids more barriers)
        float cf[NB];
        {
            float mx = s_cr[r][0];
            for (int k = 1; k < NB; ++k) mx = fmaxf(mx, s_cr[r][k]);
            float sum = 0.f;
            for (int k = 0; k < NB; ++k) { float e = __expf(s_cr[r][k] - mx); cf[k] = e; sum += e; }
            float inv = 1.f / sum;
            for (int k = 0; k < NB; ++k) cf[k] *= inv;
        }

        // banded transition: lane i owns row i of tm11/tm12/tm21/tm22 (7 diags)
        float nmu = uu, nml = ul, ncu = noise_u, ncl = noise_l, ncs = 0.f;
#pragma unroll
        for (int dd = 0; dd < 7; ++dd) {
            int j = i + dd - 3;
            int jj = j < 0 ? 0 : (j > 63 ? 63 : j);   // OOB -> basis entries are 0
            const float* p0 = s_bb + (((0 * 64 + i) * 7 + dd) * NB);
            const float* p1 = s_bb + (((1 * 64 + i) * 7 + dd) * NB);
            const float* p2 = s_bb + (((2 * 64 + i) * 7 + dd) * NB);
            const float* p3 = s_bb + (((3 * 64 + i) * 7 + dd) * NB);
            float t11 = 0.f, t12 = 0.f, t21 = 0.f, t22 = 0.f;
#pragma unroll
            for (int k = 0; k < NB; ++k) {
                t11 += cf[k] * p0[k];
                t12 += cf[k] * p1[k];
                t21 += cf[k] * p2[k];
                t22 += cf[k] * p3[k];
            }
            if (dd == 3) { t11 += 1.f; t22 += 1.f; }
            float pu = s_pm[r][jj], pl = s_pm[r][64 + jj];
            float du = s_cu[r][jj], dl = s_cl[r][jj], dsv = s_cs[r][jj];
            nmu += t11 * pu + t12 * pl;
            nml += t21 * pu + t22 * pl;
            ncu += t11 * t11 * du + 2.f * t11 * t12 * dsv + t12 * t12 * dl;
            ncl += t21 * t21 * du + 2.f * t21 * t22 * dsv + t22 * t22 * dl;
            ncs += t21 * t11 * du + (t22 * t11 + t21 * t12) * dsv + t22 * t12 * dl;
        }

        pm_out[rt * 128 + i] = nmu;       pm_out[rt * 128 + 64 + i] = nml;
        pc_out[rt * 192 + i] = ncu;       pc_out[rt * 192 + 64 + i] = ncl;
        pc_out[rt * 192 + 128 + i] = ncs;

        mu_u = nmu; mu_l = nml; cu = ncu; cl = ncl; cs = ncs;
        __syncthreads();  // protect s_pm/s_cu.. and s_in[t&1] reuse next iters
    }
}

// -------------------- kernel C: decoder MLPs --------------------------------
__global__ __launch_bounds__(32) void dec_kernel(
    const float* __restrict__ pm, const float* __restrict__ pc,
    const unsigned* wfM0, const float* bM0, const unsigned* wfM1, const float* bM1,
    const unsigned* wfM2, const float* bM2,
    const unsigned* wfV0, const float* bV0, const unsigned* wfV1, const float* bV1,
    const unsigned* wfV2, const float* bV2,
    float* __restrict__ outm, float* __restrict__ outv) {
    __shared__ unsigned sXu[16 * 192 / 2];
    __shared__ unsigned sHu[16 * 128 / 2];
    unsigned short* sX = (unsigned short*)sXu;
    unsigned short* sH = (unsigned short*)sHu;
    int row0 = blockIdx.x * 16;

    stage_tile(pm, row0, 128, 128, sX);          __syncthreads();
    layer_lds(sX, 128, wfM0, bM0, 128, sH, true); __syncthreads();
    layer_lds(sH, 128, wfM1, bM1, 128, sX, true); __syncthreads();
    layer_out(sX, 128, wfM2, bM2, 64, outm, 60, 60, row0, 0);
    __syncthreads();
    stage_tile(pc, row0, 192, 192, sX);          __syncthreads();
    layer_lds(sX, 192, wfV0, bV0, 128, sH, true); __syncthreads();
    layer_lds(sH, 128, wfV1, bV1, 128, sX, true); __syncthreads();
    layer_out(sX, 128, wfV2, bV2, 64, outv, 60, 60, row0, 2);
}

// ---------------------------------------------------------------------------
extern "C" void kernel_launch(void* const* d_in, const int* in_sizes, int n_in,
                              void* d_out, int out_size, void* d_ws, size_t ws_size,
                              hipStream_t stream) {
    const float* obs   = (const float*)d_in[0];
    const float* act   = (const float*)d_in[1];
    const unsigned char* valid = (const unsigned char*)d_in[2];
    // params flattened in dict order: enc(w,b,w,b), enc_mean, enc_var, coeff,
    // basis, ctrl(w,b,w,b), log_proc_noise, dec_mean(x3), dec_var(x3)
    const float* encW0 = (const float*)d_in[3];
    const float* encB0 = (const float*)d_in[4];
    const float* encW1 = (const float*)d_in[5];
    const float* encB1 = (const float*)d_in[6];
    const float* emW   = (const float*)d_in[7];
    const float* emB   = (const float*)d_in[8];
    const float* evW   = (const float*)d_in[9];
    const float* evB   = (const float*)d_in[10];
    const float* coW   = (const float*)d_in[11];
    const float* coB   = (const float*)d_in[12];
    const float* basis = (const float*)d_in[13];
    const float* ctW0  = (const float*)d_in[14];
    const float* ctB0  = (const float*)d_in[15];
    const float* ctW1  = (const float*)d_in[16];
    const float* ctB1  = (const float*)d_in[17];
    const float* lpn   = (const float*)d_in[18];
    const float* dmW0  = (const float*)d_in[19];
    const float* dmB0  = (const float*)d_in[20];
    const float* dmW1  = (const float*)d_in[21];
    const float* dmB1  = (const float*)d_in[22];
    const float* dmW2  = (const float*)d_in[23];
    const float* dmB2  = (const float*)d_in[24];
    const float* dvW0  = (const float*)d_in[25];
    const float* dvB0  = (const float*)d_in[26];
    const float* dvW1  = (const float*)d_in[27];
    const float* dvB1  = (const float*)d_in[28];
    const float* dvW2  = (const float*)d_in[29];
    const float* dvB2  = (const float*)d_in[30];

    // ---- workspace carve-up (floats) ----
    float* w = (float*)d_ws;
    float* obsm  = w;                 w += (size_t)BT * 64;
    float* obsv_ = w;                 w += (size_t)BT * 64;
    float* ctrlv = w;                 w += (size_t)BT * 128;
    float* pm    = w;                 w += (size_t)BT * 128;
    float* pc    = w;                 w += (size_t)BT * 192;
    float* bb    = w;                 w += BB_ELEMS;
    unsigned* uw = (unsigned*)w;
    auto carve = [&](int Kp, int Np) { unsigned* p = uw; uw += (size_t)Kp * Np / 2; return p; };
    unsigned* wfE0 = carve(64, 128);
    unsigned* wfE1 = carve(128, 128);
    unsigned* wfEM = carve(128, 64);
    unsigned* wfEV = carve(128, 64);
    unsigned* wfC0 = carve(32, 128);
    unsigned* wfC1 = carve(128, 128);
    unsigned* wfM0 = carve(128, 128);
    unsigned* wfM1 = carve(128, 128);
    unsigned* wfM2 = carve(128, 64);
    unsigned* wfV0 = carve(192, 128);
    unsigned* wfV1 = carve(128, 128);
    unsigned* wfV2 = carve(128, 64);

    float* outm = (float*)d_out;
    float* outv = outm + (size_t)BT * 60;

    // ---- weight pre-swizzle ----
    auto prep = [&](const float* W, int Ks, int Ns, int Kp, int Np, unsigned* dst) {
        int total = (Np / 16) * (Kp / 32) * 256;
        prep_bfrag<<<(total + 255) / 256, 256, 0, stream>>>(W, Ks, Ns, Kp, Np, dst);
    };
    prep(encW0, 60, 128, 64, 128, wfE0);
    prep(encW1, 128, 128, 128, 128, wfE1);
    prep(emW, 128, 64, 128, 64, wfEM);
    prep(evW, 128, 64, 128, 64, wfEV);
    prep(ctW0, 12, 128, 32, 128, wfC0);
    prep(ctW1, 128, 128, 128, 128, wfC1);
    prep(dmW0, 128, 128, 128, 128, wfM0);
    prep(dmW1, 128, 128, 128, 128, wfM1);
    prep(dmW2, 128, 60, 128, 64, wfM2);
    prep(dvW0, 192, 128, 192, 128, wfV0);
    prep(dvW1, 128, 128, 128, 128, wfV1);
    prep(dvW2, 128, 60, 128, 64, wfV2);
    prep_basis<<<(BB_ELEMS + 255) / 256, 256, 0, stream>>>(basis, bb);

    // ---- phase 1: encoder + ctrl (WMMA) ----
    enc_kernel<<<BT / 16, 32, 0, stream>>>(obs, act,
        wfE0, encB0, wfE1, encB1, wfEM, emB, wfEV, evB,
        wfC0, ctB0, wfC1, ctB1, obsm, obsv_, ctrlv);

    // ---- phase 2: sequential scan (fp32, banded, async prefetch) ----
    size_t dynBytes = (size_t)(BB_ELEMS + WC_ELEMS + 16) * sizeof(float);
    scan_kernel<<<B_DIM / 4, 256, dynBytes, stream>>>(obsm, obsv_, ctrlv, valid,
                                                      coW, coB, bb, lpn, pm, pc);

    // ---- phase 3: decoders (WMMA) ----
    dec_kernel<<<BT / 16, 32, 0, stream>>>(pm, pc,
        wfM0, dmB0, wfM1, dmB1, wfM2, dmB2,
        wfV0, dvB0, wfV1, dvB1, wfV2, dvB2, outm, outv);

    (void)in_sizes; (void)n_in; (void)out_size; (void)ws_size;
}
